// NeighborhoodAttention1D_32203664785484
// MI455X (gfx1250) — matrix operand order repaired and checked by reading, not compiled, not run on hardware
//
#include <hip/hip_runtime.h>

#define BATCH 2
#define LEN   4096
#define DIM   512
#define HEADS 8
#define KS    33
#define HD    64
#define ROWS  (BATCH * LEN)         // 8192
#define SCALE 0.125f                // HD^-0.5

typedef unsigned short u16b;
typedef __attribute__((ext_vector_type(16))) __bf16 v16bf;
typedef __attribute__((ext_vector_type(8)))  float  v8f;

union Frag { uint4 u[2]; u16b h[16]; v16bf v; };

__device__ __forceinline__ u16b f2bf(float f) {
  union { float f; unsigned int u; } un; un.f = f;
  unsigned int r = un.u + 0x7FFFu + ((un.u >> 16) & 1u);
  return (u16b)(r >> 16);
}

// ---------------- f32 -> bf16 convert ----------------
__global__ void cvt_bf16_kernel(const float* __restrict__ src,
                                u16b* __restrict__ dst, int n) {
  int i = blockIdx.x * blockDim.x + threadIdx.x;
  if (i < n) dst[i] = f2bf(src[i]);
}

// ---------------- GEMM: C[M,N] = A[M,K] * W[N,K]^T + bias ----------------
// One wave computes a 32x64 output strip: 2 A-frags x 4 B-frags -> 8 WMMA / k-step.
// mode 0: write f32 to outF (row-major MxN)
// mode 1: QKV split epilogue:
//   q,k: [(b*HEADS+h)*LEN + l]*HD + d      (row-major per token)
//   v  : [((b*HEADS+h)*HD + d)*LEN + l]    (TRANSPOSED: contiguous along L)
__global__ __launch_bounds__(256)
void gemm_bf16_kernel(const u16b* __restrict__ A, const u16b* __restrict__ W,
                      const float* __restrict__ bias, int M, int N, int K,
                      int mode, float* __restrict__ outF,
                      u16b* __restrict__ q, u16b* __restrict__ k,
                      u16b* __restrict__ v) {
  const int lane  = threadIdx.x & 31;
  const int wave  = blockIdx.x * (blockDim.x >> 5) + (threadIdx.x >> 5);
  const int ntiles = N >> 6;               // 64-wide strips
  const int mt = wave / ntiles;            // 32-row strips
  const int nt = wave % ntiles;
  if (mt >= (M >> 5)) return;
  const int m0 = mt << 5, n0 = nt << 6;
  const int mr = lane & 15, kh = lane >> 4;

  const u16b* arow0 = A + (size_t)(m0 + mr) * K;
  const u16b* arow1 = A + (size_t)(m0 + 16 + mr) * K;
  const u16b* wrow0 = W + (size_t)(n0 +  0 + mr) * K;
  const u16b* wrow1 = W + (size_t)(n0 + 16 + mr) * K;
  const u16b* wrow2 = W + (size_t)(n0 + 32 + mr) * K;
  const u16b* wrow3 = W + (size_t)(n0 + 48 + mr) * K;

  v8f acc[8] = {};

#pragma unroll 4
  for (int kb = 0; kb < K; kb += 32) {
    Frag a0, a1;
    const u16b* ap0 = arow0 + kb + kh * 16;
    const u16b* ap1 = arow1 + kb + kh * 16;
    a0.u[0] = *(const uint4*)ap0; a0.u[1] = *(const uint4*)(ap0 + 8);
    a1.u[0] = *(const uint4*)ap1; a1.u[1] = *(const uint4*)(ap1 + 8);

    Frag b0, b1, b2, b3;
    const u16b* p0 = wrow0 + kb + kh * 16;
    const u16b* p1 = wrow1 + kb + kh * 16;
    const u16b* p2 = wrow2 + kb + kh * 16;
    const u16b* p3 = wrow3 + kb + kh * 16;
    b0.u[0] = *(const uint4*)p0; b0.u[1] = *(const uint4*)(p0 + 8);
    b1.u[0] = *(const uint4*)p1; b1.u[1] = *(const uint4*)(p1 + 8);
    b2.u[0] = *(const uint4*)p2; b2.u[1] = *(const uint4*)(p2 + 8);
    b3.u[0] = *(const uint4*)p3; b3.u[1] = *(const uint4*)(p3 + 8);

    acc[0] = __builtin_amdgcn_wmma_f32_16x16x32_bf16(false, a0.v, false, b0.v, (short)0, acc[0], false, false);
    acc[1] = __builtin_amdgcn_wmma_f32_16x16x32_bf16(false, a0.v, false, b1.v, (short)0, acc[1], false, false);
    acc[2] = __builtin_amdgcn_wmma_f32_16x16x32_bf16(false, a0.v, false, b2.v, (short)0, acc[2], false, false);
    acc[3] = __builtin_amdgcn_wmma_f32_16x16x32_bf16(false, a0.v, false, b3.v, (short)0, acc[3], false, false);
    acc[4] = __builtin_amdgcn_wmma_f32_16x16x32_bf16(false, a1.v, false, b0.v, (short)0, acc[4], false, false);
    acc[5] = __builtin_amdgcn_wmma_f32_16x16x32_bf16(false, a1.v, false, b1.v, (short)0, acc[5], false, false);
    acc[6] = __builtin_amdgcn_wmma_f32_16x16x32_bf16(false, a1.v, false, b2.v, (short)0, acc[6], false, false);
    acc[7] = __builtin_amdgcn_wmma_f32_16x16x32_bf16(false, a1.v, false, b3.v, (short)0, acc[7], false, false);
  }

#pragma unroll
  for (int mt2 = 0; mt2 < 2; ++mt2) {
#pragma unroll
    for (int t = 0; t < 4; ++t) {
      const int col = n0 + t * 16 + mr;
      const float bv = bias[col];
      const v8f a = acc[mt2 * 4 + t];
#pragma unroll
      for (int r = 0; r < 8; ++r) {
        const int row = m0 + mt2 * 16 + r + 8 * kh;
        const float val = a[r] + bv;
        if (mode == 0) {
          outF[(size_t)row * N + col] = val;
        } else {
          const int which = col >> 9;
          const int h = (col >> 6) & 7;
          const int d = col & 63;
          const int b_ = row >> 12;          // row / LEN
          const int ll = row & (LEN - 1);
          const int bh = b_ * HEADS + h;
          if (which == 2) {
            // V transposed: contiguous along L for the P*V fragment loads
            v[((size_t)bh * HD + d) * LEN + ll] = f2bf(val);
          } else {
            u16b* dst = (which == 0) ? q : k;
            dst[((size_t)bh * LEN + ll) * HD + d] = f2bf(val);
          }
        }
      }
    }
  }
}

// ---------------- Neighborhood attention: one wave per (b,h,16-query tile) ----------------
// K rows staged via global_load_async_to_lds_b128 (ASYNCcnt path);
// V fragments read straight from the transposed global V buffer.
__global__ __launch_bounds__(32)
void na1d_kernel(const u16b* __restrict__ q, const u16b* __restrict__ k,
                 const u16b* __restrict__ v, u16b* __restrict__ ao) {
  __shared__ u16b  Krows[48 * 64];   // 6 KB, row-major [kk][d]
  __shared__ float Slds[16 * 48];    // 3 KB
  __shared__ u16b  Plds[16 * 64];    // 2 KB

  const int lane = threadIdx.x;            // 0..31
  const int bid  = blockIdx.x;             // (b*HEADS + h)*256 + qtile
  const int qt   = bid & 255;
  const int bh   = bid >> 8;               // 0..15
  const int l0   = qt << 4;

  int kbase = l0 - 16;
  if (kbase < 0) kbase = 0;                // 16-aligned; window union fits in 48 keys

  const u16b* qb = q + (size_t)bh * LEN * HD;
  const u16b* kb = k + (size_t)bh * LEN * HD;
  const u16b* vt = v + (size_t)bh * HD * LEN;   // transposed [d][l]

  const int mr = lane & 15, kh = lane >> 4;

  // ---- async-copy 48 K rows (6 KB) into LDS: 12 x 512B ----
  {
    const unsigned ldsK = (unsigned)(uintptr_t)(&Krows[0]);
    const int vrow = lane >> 3;            // 0..3
    const int chunk = lane & 7;            // 0..7 (16B units)
#pragma unroll
    for (int i = 0; i < 12; ++i) {
      int row = i * 4 + vrow;
      int kidx = kbase + row; if (kidx > LEN - 1) kidx = LEN - 1;
      unsigned long long ga =
          (unsigned long long)(uintptr_t)(kb + (size_t)kidx * HD + chunk * 8);
      unsigned dst = ldsK + (unsigned)(row * 128 + chunk * 16);
      asm volatile("global_load_async_to_lds_b128 %0, %1, off"
                   :: "v"(dst), "v"(ga) : "memory");
    }
  }

  // ---- load Q fragments while the async copy is in flight ----
  Frag qa[2];
#pragma unroll
  for (int c = 0; c < 2; ++c) {
    const u16b* qrow = qb + (size_t)(l0 + mr) * HD + c * 32 + kh * 16;
    qa[c].u[0] = *(const uint4*)qrow;
    qa[c].u[1] = *(const uint4*)(qrow + 8);
  }

  asm volatile("s_wait_asynccnt 0x0" ::: "memory");

  // ---- S = Q * K^T over 48 keys (3 tiles) x d=64 (2 chunks), K frags from LDS ----
  v8f s0 = {}, s1 = {}, s2 = {};
#pragma unroll
  for (int c = 0; c < 2; ++c) {
    Frag b0, b1, b2;
    const u16b* p0 = &Krows[( 0 + mr) * 64 + c * 32 + kh * 16];
    const u16b* p1 = &Krows[(16 + mr) * 64 + c * 32 + kh * 16];
    const u16b* p2 = &Krows[(32 + mr) * 64 + c * 32 + kh * 16];
    b0.u[0] = *(const uint4*)p0; b0.u[1] = *(const uint4*)(p0 + 8);
    b1.u[0] = *(const uint4*)p1; b1.u[1] = *(const uint4*)(p1 + 8);
    b2.u[0] = *(const uint4*)p2; b2.u[1] = *(const uint4*)(p2 + 8);

    s0 = __builtin_amdgcn_wmma_f32_16x16x32_bf16(false, qa[c].v, false, b0.v, (short)0, s0, false, false);
    s1 = __builtin_amdgcn_wmma_f32_16x16x32_bf16(false, qa[c].v, false, b1.v, (short)0, s1, false, false);
    s2 = __builtin_amdgcn_wmma_f32_16x16x32_bf16(false, qa[c].v, false, b2.v, (short)0, s2, false, false);
  }
  v8f ss[3] = { s0, s1, s2 };
#pragma unroll
  for (int t = 0; t < 3; ++t)
#pragma unroll
    for (int r = 0; r < 8; ++r)
      Slds[(r + 8 * kh) * 48 + t * 16 + mr] = ss[t][r] * SCALE;
  __syncthreads();

  // ---- masked row softmax (lanes 0..15, one row each) ----
  if (lane < 16) {
    const int m = lane;
    const int l = l0 + m;
    int st = l - (KS / 2);
    if (st < 0) st = 0;
    if (st > LEN - KS) st = LEN - KS;
    float vals[48];
    float mx = -1e30f;
#pragma unroll
    for (int j = 0; j < 48; ++j) {
      const int kidx = kbase + j;
      const bool ok = (kidx >= st) && (kidx < st + KS);
      const float x = ok ? Slds[m * 48 + j] : -1e30f;
      vals[j] = x;
      mx = fmaxf(mx, x);
    }
    float sum = 0.f;
#pragma unroll
    for (int j = 0; j < 48; ++j) {
      const float e = (vals[j] > -1e29f) ? __expf(vals[j] - mx) : 0.f;
      vals[j] = e;
      sum += e;
    }
    const float inv = 1.f / sum;
#pragma unroll
    for (int j = 0; j < 48; ++j) Plds[m * 64 + j] = f2bf(vals[j] * inv);
#pragma unroll
    for (int j = 48; j < 64; ++j) Plds[m * 64 + j] = 0;
  }
  __syncthreads();

  // ---- O = P * V : P frags from LDS, V frags straight from transposed global V ----
  v8f o0 = {}, o1 = {}, o2 = {}, o3 = {};
#pragma unroll
  for (int c = 0; c < 2; ++c) {
    Frag a;
    const u16b* prow = &Plds[mr * 64 + c * 32 + kh * 16];
    a.u[0] = *(const uint4*)prow;
    a.u[1] = *(const uint4*)(prow + 8);

    // b[i] = V[k = kbase + c*32 + koff + i][n = t*16 + mr]  (contiguous along L)
    Frag b0, b1, b2, b3;
    const u16b* v0 = vt + ((size_t)( 0 + mr)) * LEN + kbase + c * 32 + kh * 16;
    const u16b* v1 = vt + ((size_t)(16 + mr)) * LEN + kbase + c * 32 + kh * 16;
    const u16b* v2 = vt + ((size_t)(32 + mr)) * LEN + kbase + c * 32 + kh * 16;
    const u16b* v3 = vt + ((size_t)(48 + mr)) * LEN + kbase + c * 32 + kh * 16;
    b0.u[0] = *(const uint4*)v0; b0.u[1] = *(const uint4*)(v0 + 8);
    b1.u[0] = *(const uint4*)v1; b1.u[1] = *(const uint4*)(v1 + 8);
    b2.u[0] = *(const uint4*)v2; b2.u[1] = *(const uint4*)(v2 + 8);
    b3.u[0] = *(const uint4*)v3; b3.u[1] = *(const uint4*)(v3 + 8);

    o0 = __builtin_amdgcn_wmma_f32_16x16x32_bf16(false, a.v, false, b0.v, (short)0, o0, false, false);
    o1 = __builtin_amdgcn_wmma_f32_16x16x32_bf16(false, a.v, false, b1.v, (short)0, o1, false, false);
    o2 = __builtin_amdgcn_wmma_f32_16x16x32_bf16(false, a.v, false, b2.v, (short)0, o2, false, false);
    o3 = __builtin_amdgcn_wmma_f32_16x16x32_bf16(false, a.v, false, b3.v, (short)0, o3, false, false);
  }
  v8f oo[4] = { o0, o1, o2, o3 };
  const int b_ = bh >> 3, h = bh & 7;
#pragma unroll
  for (int t = 0; t < 4; ++t)
#pragma unroll
    for (int r = 0; r < 8; ++r) {
      const int row = l0 + r + 8 * kh;
      const int d = t * 16 + mr;
      ao[((size_t)(b_ * LEN + row)) * DIM + h * HD + d] = f2bf(oo[t][r]);
    }
}

extern "C" void kernel_launch(void* const* d_in, const int* in_sizes, int n_in,
                              void* d_out, int out_size, void* d_ws, size_t ws_size,
                              hipStream_t stream) {
  const float* x      = (const float*)d_in[0];
  const float* w_qkv  = (const float*)d_in[1];
  const float* b_qkv  = (const float*)d_in[2];
  const float* w_proj = (const float*)d_in[3];
  const float* b_proj = (const float*)d_in[4];
  float* out = (float*)d_out;

  char* ws = (char*)d_ws;
  size_t off = 0;
  auto alloc = [&](size_t bytes) -> void* {
    void* p = ws + off;
    off += (bytes + 255) & ~(size_t)255;
    return p;
  };
  u16b* x_bf     = (u16b*)alloc((size_t)ROWS * DIM * 2);        // 8 MB
  u16b* wqkv_bf  = (u16b*)alloc((size_t)3 * DIM * DIM * 2);     // 1.5 MB
  u16b* wproj_bf = (u16b*)alloc((size_t)DIM * DIM * 2);         // 0.5 MB
  u16b* qs       = (u16b*)alloc((size_t)BATCH * HEADS * LEN * HD * 2);
  u16b* ks       = (u16b*)alloc((size_t)BATCH * HEADS * LEN * HD * 2);
  u16b* vs       = (u16b*)alloc((size_t)BATCH * HEADS * LEN * HD * 2); // transposed
  u16b* ao       = (u16b*)alloc((size_t)ROWS * DIM * 2);        // 8 MB

  // 1) f32 -> bf16 conversions
  {
    int n = ROWS * DIM;
    cvt_bf16_kernel<<<(n + 255) / 256, 256, 0, stream>>>(x, x_bf, n);
    n = 3 * DIM * DIM;
    cvt_bf16_kernel<<<(n + 255) / 256, 256, 0, stream>>>(w_qkv, wqkv_bf, n);
    n = DIM * DIM;
    cvt_bf16_kernel<<<(n + 255) / 256, 256, 0, stream>>>(w_proj, wproj_bf, n);
  }

  // 2) QKV GEMM: [8192,512] x [1536,512]^T, split epilogue (V transposed)
  {
    int waves = (ROWS / 32) * ((3 * DIM) / 64);   // 256 * 24 = 6144
    gemm_bf16_kernel<<<waves / 8, 256, 0, stream>>>(
        x_bf, wqkv_bf, b_qkv, ROWS, 3 * DIM, DIM, 1, nullptr, qs, ks, vs);
  }

  // 3) Neighborhood attention: one wave per (b,h,16-query tile)
  na1d_kernel<<<BATCH * HEADS * (LEN / 16), 32, 0, stream>>>(qs, ks, vs, ao);

  // 4) Output projection: [8192,512] x [512,512]^T -> f32 out
  {
    int waves = (ROWS / 32) * (DIM / 64);         // 256 * 8 = 2048
    gemm_bf16_kernel<<<waves / 8, 256, 0, stream>>>(
        ao, wproj_bf, b_proj, ROWS, DIM, DIM, 0, out, nullptr, nullptr, nullptr);
  }
}
